// GRUEncoder_20151986552857
// MI455X (gfx1250) — matrix operand order repaired and checked by reading, not compiled, not run on hardware
//
#include <hip/hip_runtime.h>

// ---------------------------------------------------------------------------
// GRU encoder for MI455X (gfx1250, wave32, WMMA bf16 16x16x32, async->LDS)
// ---------------------------------------------------------------------------
#define Bk   32
#define Tk   1024
#define Ik   256
#define Hk   512
#define Lk   128
#define G3H  1536   // 3*H
#define NWG  32     // persistent workgroups in recurrent kernel

typedef __attribute__((ext_vector_type(16))) __bf16 v16bf;
typedef __attribute__((ext_vector_type(8)))  __bf16 v8bf;
typedef __attribute__((ext_vector_type(8)))  float  v8f;

__device__ __forceinline__ unsigned short f2bf(float f) {
    unsigned u = __float_as_uint(f);
    u += 0x7FFFu + ((u >> 16) & 1u);           // round-to-nearest-even
    return (unsigned short)(u >> 16);
}
__device__ __forceinline__ float bf2f(unsigned short h) {
    return __uint_as_float(((unsigned)h) << 16);
}
__device__ __forceinline__ float sigmoidf(float x) {
    return 1.0f / (1.0f + __expf(-x));
}

// A fragment (16x32 bf16, MxK): lane holds row m = mbase + (lane&15),
// K halves {0..7,16..23} (lanes 0-15) / {8..15,24..31} (lanes 16-31).
__device__ __forceinline__ v16bf frag_a(const unsigned short* A, int lda, int mbase, int kk) {
    const int lane = threadIdx.x & 31;
    const int m  = mbase + (lane & 15);
    const int k0 = (lane >> 4) * 8;
    const unsigned short* p = A + (size_t)m * lda + kk * 32 + k0;
    v8bf lo = *(const v8bf*)p;
    v8bf hi = *(const v8bf*)(p + 16);
    v16bf r;
#pragma unroll
    for (int i = 0; i < 8; ++i) { r[i] = lo[i]; r[i + 8] = hi[i]; }
    return r;
}
// B fragment (32x16 bf16, KxN) for D = A * W^T: B[k,n] = W[n,k].
// lane holds column n = nbase + (lane&15), K halves {0..7,16..23}
// (lanes 0-15) / {8..15,24..31} (lanes 16-31) => two 16B loads of row n of W.
__device__ __forceinline__ v16bf frag_b(const unsigned short* W, int ldw, int nbase, int kk) {
    const int lane = threadIdx.x & 31;
    const int n  = nbase + (lane & 15);
    const int kb = (lane >> 4) * 8;
    const unsigned short* p = W + (size_t)n * ldw + kk * 32 + kb;
    v8bf lo = *(const v8bf*)p;
    v8bf hi = *(const v8bf*)(p + 16);
    v16bf r;
#pragma unroll
    for (int i = 0; i < 8; ++i) { r[i] = lo[i]; r[i + 8] = hi[i]; }
    return r;
}
__device__ __forceinline__ v8f wmma_bf16(v16bf a, v16bf b, v8f c) {
    return __builtin_amdgcn_wmma_f32_16x16x32_bf16(false, a, false, b, (short)0, c, false, false);
}

// ---------------------------------------------------------------------------
// K0: fp32 -> bf16 conversion
// ---------------------------------------------------------------------------
__global__ void __launch_bounds__(256) k_cvt(const float* __restrict__ s,
                                             unsigned short* __restrict__ d, int n) {
    int i = blockIdx.x * 256 + threadIdx.x;
    if (i < n) d[i] = f2bf(s[i]);
}

// ---------------------------------------------------------------------------
// K1: fused input_proj (WMMA) + LayerNorm.  One block = 16-row strip of h.
// ---------------------------------------------------------------------------
__global__ void __launch_bounds__(256) k_proj_ln(const unsigned short* __restrict__ xb,
                                                 const unsigned short* __restrict__ winb,
                                                 const float* __restrict__ b_in,
                                                 const float* __restrict__ gamma,
                                                 const float* __restrict__ beta,
                                                 unsigned short* __restrict__ hn) {
    __shared__ __align__(16) float hlds[16][Hk];
    const int wave = threadIdx.x >> 5;
    const int lane = threadIdx.x & 31;
    const int m0   = blockIdx.x * 16;

    // GEMM: 32 N-tiles, 4 per wave (shared A fragment), K = 256 (8 k-steps)
    {
        const int n0 = wave * 64;
        v8f acc[4];
#pragma unroll
        for (int q = 0; q < 4; ++q) acc[q] = (v8f){0.f,0.f,0.f,0.f,0.f,0.f,0.f,0.f};
#pragma unroll
        for (int kk = 0; kk < 8; ++kk) {
            v16bf a = frag_a(xb, Ik, m0, kk);
#pragma unroll
            for (int q = 0; q < 4; ++q)
                acc[q] = wmma_bf16(a, frag_b(winb, Ik, n0 + q * 16, kk), acc[q]);
        }
#pragma unroll
        for (int q = 0; q < 4; ++q) {
            const int col  = n0 + q * 16 + (lane & 15);
            const float bi = b_in[col];
#pragma unroll
            for (int j = 0; j < 8; ++j)
                hlds[j + ((lane >> 4) << 3)][col] = acc[q][j] + bi;
        }
    }
    __syncthreads();

    // LayerNorm: 2 rows per wave, wave32 shuffle reduction
    const float inv = 1.0f / (float)Hk;
    for (int rr = 0; rr < 2; ++rr) {
        const int r = wave * 2 + rr;
        float v[16], s = 0.f, sq = 0.f;
#pragma unroll
        for (int i = 0; i < 16; ++i) {
            float x = hlds[r][lane + 32 * i];
            v[i] = x; s += x; sq += x * x;
        }
#pragma unroll
        for (int off = 16; off >= 1; off >>= 1) {
            s  += __shfl_xor(s,  off, 32);
            sq += __shfl_xor(sq, off, 32);
        }
        const float mean = s * inv;
        const float rstd = rsqrtf(sq * inv - mean * mean + 1e-5f);
        const size_t ro = (size_t)(m0 + r) * Hk;
#pragma unroll
        for (int i = 0; i < 16; ++i) {
            const int c = lane + 32 * i;
            hn[ro + c] = f2bf((v[i] - mean) * rstd * gamma[c] + beta[c]);
        }
    }
}

// ---------------------------------------------------------------------------
// K2: gi = hn @ w_ih^T + b_ih   (M=32768, N=1536, K=512) -> bf16
// one wave per 16x64 group (4 tiles share one A fragment per k-step)
// ---------------------------------------------------------------------------
__global__ void __launch_bounds__(256) k_gi(const unsigned short* __restrict__ hn,
                                            const unsigned short* __restrict__ wihb,
                                            const float* __restrict__ b_ih,
                                            unsigned short* __restrict__ gi) {
    const int wid  = blockIdx.x * 8 + (threadIdx.x >> 5);   // group id, 2048*24 total
    const int lane = threadIdx.x & 31;
    const int m0   = (wid / 24) * 16;
    const int n0   = (wid % 24) * 64;
    v8f acc[4];
#pragma unroll
    for (int q = 0; q < 4; ++q) acc[q] = (v8f){0.f,0.f,0.f,0.f,0.f,0.f,0.f,0.f};
#pragma unroll
    for (int kk = 0; kk < 16; ++kk) {
        v16bf a = frag_a(hn, Hk, m0, kk);
#pragma unroll
        for (int q = 0; q < 4; ++q)
            acc[q] = wmma_bf16(a, frag_b(wihb, Hk, n0 + q * 16, kk), acc[q]);
    }
#pragma unroll
    for (int q = 0; q < 4; ++q) {
        const int col  = n0 + q * 16 + (lane & 15);
        const float bi = b_ih[col];
#pragma unroll
        for (int j = 0; j < 8; ++j) {
            const size_t row = (size_t)(m0 + j + ((lane >> 4) << 3));
            gi[row * G3H + col] = f2bf(acc[q][j] + bi);
        }
    }
}

// ---------------------------------------------------------------------------
// K3: recurrent GRU scan. 32 persistent WGs, WG g owns h columns [g*16,g*16+16).
// w_hh slice (48x512 bf16 = 48KB) brought into LDS via GLOBAL_LOAD_ASYNC_TO_LDS
// (ASYNCcnt path); h double-buffered in global (bf16 for WMMA A, fp32 master);
// one device-scope barrier per step; gi for step t+1 prefetched.
// ---------------------------------------------------------------------------
__global__ void __launch_bounds__(256) k_gru(const unsigned short* __restrict__ gi,
                                             const unsigned short* __restrict__ whhb,
                                             const float* __restrict__ b_hh,
                                             float* __restrict__ h32,
                                             unsigned short* __restrict__ hb16,
                                             unsigned* __restrict__ bar) {
    __shared__ __align__(32) unsigned short wlds[48 * Hk];   // 48KB
    __shared__ __align__(16) float ghlds[32][48];            // 6KB
    const int tid  = threadIdx.x;
    const int lane = tid & 31;
    const int wave = tid >> 5;
    const int wg   = blockIdx.x;              // 0..31

    // preload this WG's w_hh slice with CDNA5 async global->LDS copies
    // (16B per lane per issue, tracked by ASYNCcnt)
    for (int g = 0; g < 3; ++g) {
        const unsigned short* src = whhb + ((size_t)g * Hk + wg * 16) * Hk;
        unsigned short* dst = wlds + g * 16 * Hk;
        for (int i = tid; i < 1024; i += 256) {
            unsigned long long ga = (unsigned long long)(uintptr_t)(src + i * 8);
            unsigned la = (unsigned)(uintptr_t)(dst + i * 8);  // low 32b = LDS offset
            asm volatile("global_load_async_to_lds_b128 %0, %1, off"
                         :: "v"(la), "v"(ga) : "memory");
        }
    }
    asm volatile("s_wait_asynccnt 0" ::: "memory");
    __syncthreads();

    const int mi = wave & 1;        // m-tile (rows 0-15 / 16-31)
    const int nt = wave >> 1;       // gate tile (r/z/n), valid for wave<6

    for (int t = 0; t < Tk; ++t) {
        const unsigned short* hbuf = hb16 + (size_t)(t & 1) * (Bk * Hk);
        unsigned short* hnext      = hb16 + (size_t)((t + 1) & 1) * (Bk * Hk);

        if (wave < 6) {             // gh = h_prev @ w_hh_slice^T  (32x48)
            v8f acc = {0.f, 0.f, 0.f, 0.f, 0.f, 0.f, 0.f, 0.f};
#pragma unroll
            for (int kk = 0; kk < 16; ++kk)
                acc = wmma_bf16(frag_a(hbuf, Hk, mi * 16, kk),
                                frag_b(wlds, Hk, nt * 16, kk), acc);
            const int cl = nt * 16 + (lane & 15);
#pragma unroll
            for (int j = 0; j < 8; ++j)
                ghlds[mi * 16 + j + ((lane >> 4) << 3)][cl] = acc[j];
        }
        __syncthreads();

        // gates + state update for own 16 columns, all 32 batch rows
        for (int e = tid; e < 512; e += 256) {
            const int b = e >> 4, c = e & 15;
            const int col = wg * 16 + c;
            const float ghr = ghlds[b][c]      + b_hh[col];
            const float ghz = ghlds[b][16 + c] + b_hh[Hk + col];
            const float ghn = ghlds[b][32 + c] + b_hh[2 * Hk + col];
            const size_t gb = ((size_t)b * Tk + t) * G3H;
            const float r = sigmoidf(bf2f(gi[gb + col])          + ghr);
            const float z = sigmoidf(bf2f(gi[gb + Hk + col])     + ghz);
            const float n = tanhf  (bf2f(gi[gb + 2 * Hk + col])  + r * ghn);
            const float hp = h32[(size_t)b * Hk + col];
            const float hv = (1.0f - z) * n + z * hp;
            h32[(size_t)b * Hk + col]  = hv;
            hnext[(size_t)b * Hk + col] = f2bf(hv);
            // prefetch next step's gate-input lines (global_prefetch_b8)
            if (c == 0 && t + 1 < Tk) {
                const unsigned short* nb = gi + gb + G3H + wg * 16;
                __builtin_prefetch(nb, 0, 0);
                __builtin_prefetch(nb + Hk, 0, 0);
                __builtin_prefetch(nb + 2 * Hk, 0, 0);
            }
        }
        __syncthreads();

        // device-scope generation barrier across 32 WGs
        if (tid == 0) {
            __threadfence();
            unsigned arrived = __hip_atomic_fetch_add(bar, 1u, __ATOMIC_ACQ_REL,
                                                      __HIP_MEMORY_SCOPE_AGENT) + 1;
            if (arrived == NWG) {
                __hip_atomic_store(bar, 0u, __ATOMIC_RELAXED, __HIP_MEMORY_SCOPE_AGENT);
                __hip_atomic_store(bar + 1, (unsigned)(t + 1), __ATOMIC_RELEASE,
                                   __HIP_MEMORY_SCOPE_AGENT);
            } else {
                while (__hip_atomic_load(bar + 1, __ATOMIC_ACQUIRE,
                                         __HIP_MEMORY_SCOPE_AGENT) <= (unsigned)t)
                    __builtin_amdgcn_s_sleep(1);
            }
        }
        __syncthreads();
    }
}

// ---------------------------------------------------------------------------
// K4: latent = h_last @ w_lat^T + b_lat ; copy hidden into output
// out layout: [0,4096) latent [B,L] ; [4096,20480) hidden [1,B,H]
// ---------------------------------------------------------------------------
__global__ void __launch_bounds__(256) k_lat(const float* __restrict__ h32,
                                             const float* __restrict__ wlat,
                                             const float* __restrict__ blat,
                                             float* __restrict__ out) {
    const int i = blockIdx.x * 256 + threadIdx.x;
    if (i < Bk * Lk) {
        const int b = i >> 7, l = i & (Lk - 1);
        const float* hr = h32 + (size_t)b * Hk;
        const float* wr = wlat + (size_t)l * Hk;
        float s = blat[l];
        for (int k = 0; k < Hk; ++k) s += hr[k] * wr[k];
        out[i] = s;
    } else if (i < Bk * Lk + Bk * Hk) {
        out[i] = h32[i - Bk * Lk];
    }
}

// ---------------------------------------------------------------------------
extern "C" void kernel_launch(void* const* d_in, const int* in_sizes, int n_in,
                              void* d_out, int out_size, void* d_ws, size_t ws_size,
                              hipStream_t stream) {
    (void)in_sizes; (void)n_in; (void)out_size; (void)ws_size;
    const float* x     = (const float*)d_in[0];
    const float* w_in  = (const float*)d_in[1];
    const float* b_in  = (const float*)d_in[2];
    const float* gamma = (const float*)d_in[3];
    const float* beta  = (const float*)d_in[4];
    const float* w_ih  = (const float*)d_in[5];
    const float* w_hh  = (const float*)d_in[6];
    const float* b_ih  = (const float*)d_in[7];
    const float* b_hh  = (const float*)d_in[8];
    const float* w_lat = (const float*)d_in[9];
    const float* b_lat = (const float*)d_in[10];

    char* ws = (char*)d_ws;
    // workspace layout (256B aligned)
    unsigned short* xb   = (unsigned short*)(ws + 0);            // 16,777,216 B
    unsigned short* winb = (unsigned short*)(ws + 16777216);     //    262,144 B
    unsigned short* wihb = (unsigned short*)(ws + 17039360);     //  1,572,864 B
    unsigned short* whhb = (unsigned short*)(ws + 18612224);     //  1,572,864 B
    unsigned short* hn   = (unsigned short*)(ws + 20185088);     // 33,554,432 B
    unsigned short* gib  = (unsigned short*)(ws + 53739520);     //100,663,296 B
    float*          h32  = (float*)         (ws + 154402816);    //     65,536 B
    unsigned short* hb16 = (unsigned short*)(ws + 154468352);    //     65,536 B (x2 buf)
    unsigned*       bar  = (unsigned*)      (ws + 154533888);    //        256 B

    // fp32 -> bf16 conversions
    k_cvt<<<(Bk * Tk * Ik + 255) / 256, 256, 0, stream>>>(x, xb, Bk * Tk * Ik);
    k_cvt<<<(Hk * Ik + 255) / 256,      256, 0, stream>>>(w_in, winb, Hk * Ik);
    k_cvt<<<(G3H * Hk + 255) / 256,     256, 0, stream>>>(w_ih, wihb, G3H * Hk);
    k_cvt<<<(G3H * Hk + 255) / 256,     256, 0, stream>>>(w_hh, whhb, G3H * Hk);

    // fused proj + LayerNorm (2048 strips of 16 rows)
    k_proj_ln<<<(Bk * Tk) / 16, 256, 0, stream>>>(xb, winb, b_in, gamma, beta, hn);

    // gate-input GEMM: 49152 wave-groups of 4 tiles / 8 waves per block
    k_gi<<<(2048 * 24) / 8, 256, 0, stream>>>(hn, wihb, b_ih, gib);

    // zero h state (both buffers) + barrier, then run the recurrent scan
    (void)hipMemsetAsync(h32, 0, 65536 + 65536 + 256, stream);
    k_gru<<<NWG, 256, 0, stream>>>(gib, whhb, b_hh, h32, hb16, bar);

    // latent projection + hidden copy
    k_lat<<<(Bk * Lk + Bk * Hk + 255) / 256, 256, 0, stream>>>(h32, w_lat, b_lat,
                                                               (float*)d_out);
}
